// MoELayer_62732292326119
// MI455X (gfx1250) — matrix operand order repaired and checked by reading, not compiled, not run on hardware
//
#include <hip/hip_runtime.h>
#include <hip/hip_bf16.h>

typedef __bf16 bf16_t;
typedef bf16_t v16bf __attribute__((ext_vector_type(16)));
typedef bf16_t v8bf  __attribute__((ext_vector_type(8)));
typedef float  v8f   __attribute__((ext_vector_type(8)));

// Problem dims from the reference
constexpr int B_ = 4, S_ = 2048, D_ = 1024, H_ = 2048, E_ = 8;
constexpr int NT = B_ * S_;     // 8192 tokens
constexpr float NEG_SLOPE = 0.01f;

constexpr int BM = 128, BN = 128, KT = 32;
constexpr int LDK = 40;         // padded LDS row stride in bf16 elems (80B, 16B-aligned)

#if __has_builtin(__builtin_amdgcn_global_load_async_to_lds_b128)
#define USE_ASYNC_LDS 1
#else
#define USE_ASYNC_LDS 0
#endif

// Types matching the builtin's prototype: vector_size(16) int, AS1 (global) / AS3 (LDS)
typedef int v4i_ __attribute__((vector_size(16)));
typedef __attribute__((address_space(1))) v4i_* gp4;
typedef __attribute__((address_space(3))) v4i_* lp4;

__device__ __forceinline__ void wait_async0() {
#if __has_builtin(__builtin_amdgcn_s_wait_asynccnt)
  __builtin_amdgcn_s_wait_asynccnt(0);
#else
  asm volatile("s_wait_asynccnt 0" ::: "memory");
#endif
}

// ---------------- f32 -> bf16 (8 elems / thread) ----------------
__global__ void cvt_f32_bf16(const float* __restrict__ in, bf16_t* __restrict__ out, long n) {
  long i = ((long)blockIdx.x * blockDim.x + threadIdx.x) * 8;
  if (i >= n) return;
  const float4* p = (const float4*)(in + i);
  float4 a = p[0], b = p[1];
  v8bf v;
  v[0] = (bf16_t)a.x; v[1] = (bf16_t)a.y; v[2] = (bf16_t)a.z; v[3] = (bf16_t)a.w;
  v[4] = (bf16_t)b.x; v[5] = (bf16_t)b.y; v[6] = (bf16_t)b.z; v[7] = (bf16_t)b.w;
  *(v8bf*)(out + i) = v;
}

// ---------------- gate logits + softmax (f32 exact) ----------------
// 256 threads = 32 tokens/block, 8 lanes per token (one expert each).
__global__ void gate_kernel(const float* __restrict__ X, const float* __restrict__ Wg,
                            const float* __restrict__ bg, float* __restrict__ gateOut) {
  int t = threadIdx.x;
  int e = t & 7;
  int token = blockIdx.x * 32 + (t >> 3);
  const float4* xr = (const float4*)(X + (size_t)token * D_);
  const float4* wr = (const float4*)(Wg + (size_t)e * D_);
  float s = 0.f;
  for (int i = 0; i < D_ / 4; ++i) {
    float4 a = xr[i], b = wr[i];
    s += a.x * b.x + a.y * b.y + a.z * b.z + a.w * b.w;
  }
  s += bg[e];
  float m = s;
  for (int off = 1; off < 8; off <<= 1) m = fmaxf(m, __shfl_xor(m, off, 8));
  float ex = __expf(s - m);
  float sum = ex;
  for (int off = 1; off < 8; off <<= 1) sum += __shfl_xor(sum, off, 8);
  gateOut[(size_t)token * E_ + e] = ex / sum;
}

// ---------------- fused MoE GEMM: loops experts in-register ----------------
__global__ __launch_bounds__(256)
void moe_wmma(const bf16_t* __restrict__ Xb, const bf16_t* __restrict__ Web,
              const float* __restrict__ be, const float* __restrict__ gate,
              float* __restrict__ out) {
  __shared__ bf16_t aT[2][BM * LDK];  // 2 x 10 KB (double buffered)
  __shared__ bf16_t bT[2][BN * LDK];  // 2 x 10 KB
  __shared__ float  gT[BM * E_];      // 4 KB gate tile

  const int t  = threadIdx.x;
  const int m0 = blockIdx.y * BM;     // token tile
  const int n0 = blockIdx.x * BN;     // hidden tile

  // stage gate probs for this token tile (contiguous 1024 floats)
  ((float4*)gT)[t] = ((const float4*)(gate + (size_t)m0 * E_))[t];

  const int lane   = t & 31, wave = t >> 5;
  const int waveM  = wave & 3;        // M offset = waveM*32
  const int waveN  = wave >> 2;       // N offset = waveN*64
  const int laneLo = lane & 15, laneHi = lane >> 4;

  // per-thread staging chunks: 2 for A, 2 for B (16B each)
  const int c0 = t,       rA0 = c0 >> 2, cA0 = (c0 & 3) * 8;
  const int c1 = t + 256, rA1 = c1 >> 2, cA1 = (c1 & 3) * 8;

  v8f fin[2][4] = {};                 // expert-summed output accumulators

  #pragma unroll 1
  for (int e = 0; e < E_; ++e) {
    const bf16_t* Wb = Web + (size_t)e * H_ * D_;

    float bias[4];
    #pragma unroll
    for (int ni = 0; ni < 4; ++ni)
      bias[ni] = be[e * H_ + n0 + waveN * 64 + ni * 16 + laneLo];

    v8f acc[2][4] = {};

#if USE_ASYNC_LDS
    // issue async loads of k-tile 0 into buffer 0 (ASYNCcnt-tracked, no VGPR staging)
    __builtin_amdgcn_global_load_async_to_lds_b128(
        (gp4)(Xb + (size_t)(m0 + rA0) * D_ + cA0), (lp4)(aT[0] + rA0 * LDK + cA0), 0, 0);
    __builtin_amdgcn_global_load_async_to_lds_b128(
        (gp4)(Xb + (size_t)(m0 + rA1) * D_ + cA1), (lp4)(aT[0] + rA1 * LDK + cA1), 0, 0);
    __builtin_amdgcn_global_load_async_to_lds_b128(
        (gp4)(Wb + (size_t)(n0 + rA0) * D_ + cA0), (lp4)(bT[0] + rA0 * LDK + cA0), 0, 0);
    __builtin_amdgcn_global_load_async_to_lds_b128(
        (gp4)(Wb + (size_t)(n0 + rA1) * D_ + cA1), (lp4)(bT[0] + rA1 * LDK + cA1), 0, 0);
#else
    uint4 aR[2], bR[2];
    aR[0] = *(const uint4*)(Xb + (size_t)(m0 + rA0) * D_ + cA0);
    aR[1] = *(const uint4*)(Xb + (size_t)(m0 + rA1) * D_ + cA1);
    bR[0] = *(const uint4*)(Wb + (size_t)(n0 + rA0) * D_ + cA0);
    bR[1] = *(const uint4*)(Wb + (size_t)(n0 + rA1) * D_ + cA1);
    *(uint4*)(aT[0] + rA0 * LDK + cA0) = aR[0];
    *(uint4*)(aT[0] + rA1 * LDK + cA1) = aR[1];
    *(uint4*)(bT[0] + rA0 * LDK + cA0) = bR[0];
    *(uint4*)(bT[0] + rA1 * LDK + cA1) = bR[1];
#endif

    for (int kt = 0; kt < D_ / KT; ++kt) {
      const int cur = kt & 1;

#if USE_ASYNC_LDS
      wait_async0();          // my tile-kt async loads landed in LDS
      __syncthreads();        // everyone's landed; everyone done reading buf[1-cur]
      if (kt + 1 < D_ / KT) { // prefetch tile kt+1 straight into the other buffer
        const size_t k0 = (size_t)(kt + 1) * KT;
        __builtin_amdgcn_global_load_async_to_lds_b128(
            (gp4)(Xb + (size_t)(m0 + rA0) * D_ + k0 + cA0), (lp4)(aT[1 - cur] + rA0 * LDK + cA0), 0, 0);
        __builtin_amdgcn_global_load_async_to_lds_b128(
            (gp4)(Xb + (size_t)(m0 + rA1) * D_ + k0 + cA1), (lp4)(aT[1 - cur] + rA1 * LDK + cA1), 0, 0);
        __builtin_amdgcn_global_load_async_to_lds_b128(
            (gp4)(Wb + (size_t)(n0 + rA0) * D_ + k0 + cA0), (lp4)(bT[1 - cur] + rA0 * LDK + cA0), 0, 0);
        __builtin_amdgcn_global_load_async_to_lds_b128(
            (gp4)(Wb + (size_t)(n0 + rA1) * D_ + k0 + cA1), (lp4)(bT[1 - cur] + rA1 * LDK + cA1), 0, 0);
      }
#else
      __syncthreads();        // buf[cur] complete; prior reads of buf[1-cur] done
      uint4 aR[2], bR[2];
      if (kt + 1 < D_ / KT) {
        const size_t k0 = (size_t)(kt + 1) * KT;
        aR[0] = *(const uint4*)(Xb + (size_t)(m0 + rA0) * D_ + k0 + cA0);
        aR[1] = *(const uint4*)(Xb + (size_t)(m0 + rA1) * D_ + k0 + cA1);
        bR[0] = *(const uint4*)(Wb + (size_t)(n0 + rA0) * D_ + k0 + cA0);
        bR[1] = *(const uint4*)(Wb + (size_t)(n0 + rA1) * D_ + k0 + cA1);
      }
#endif

      // A fragments: lanes 0-15 -> K{0..7,16..23}, lanes 16-31 -> K{8..15,24..31}
      v16bf af[2], bfr[4];
      #pragma unroll
      for (int mi = 0; mi < 2; ++mi) {
        const bf16_t* p = aT[cur] + (waveM * 32 + mi * 16 + laneLo) * LDK + laneHi * 8;
        v8bf lo = *(const v8bf*)p;
        v8bf hi = *(const v8bf*)(p + 16);
        #pragma unroll
        for (int j = 0; j < 8; ++j) { af[mi][j] = lo[j]; af[mi][j + 8] = hi[j]; }
      }
      // B fragments: same K order as A (lanes 0-15 -> K{0..7,16..23}, lanes 16-31 -> K{8..15,24..31})
      #pragma unroll
      for (int ni = 0; ni < 4; ++ni) {
        const bf16_t* p = bT[cur] + (waveN * 64 + ni * 16 + laneLo) * LDK + laneHi * 8;
        v8bf lo = *(const v8bf*)p;
        v8bf hi = *(const v8bf*)(p + 16);
        #pragma unroll
        for (int j = 0; j < 8; ++j) { bfr[ni][j] = lo[j]; bfr[ni][j + 8] = hi[j]; }
      }

      #pragma unroll
      for (int mi = 0; mi < 2; ++mi)
        #pragma unroll
        for (int ni = 0; ni < 4; ++ni)
          acc[mi][ni] = __builtin_amdgcn_wmma_f32_16x16x32_bf16(
              false, af[mi], false, bfr[ni], (short)0, acc[mi][ni], false, false);

#if !USE_ASYNC_LDS
      if (kt + 1 < D_ / KT) {   // store prefetched tile into the other buffer
        *(uint4*)(aT[1 - cur] + rA0 * LDK + cA0) = aR[0];
        *(uint4*)(aT[1 - cur] + rA1 * LDK + cA1) = aR[1];
        *(uint4*)(bT[1 - cur] + rA0 * LDK + cA0) = bR[0];
        *(uint4*)(bT[1 - cur] + rA1 * LDK + cA1) = bR[1];
      }
#endif
    }

    // epilogue: bias + LeakyReLU + gate-weighted accumulate
    #pragma unroll
    for (int mi = 0; mi < 2; ++mi)
      #pragma unroll
      for (int ni = 0; ni < 4; ++ni)
        #pragma unroll
        for (int i = 0; i < 8; ++i) {
          int rloc = waveM * 32 + mi * 16 + laneHi * 8 + i;   // C layout: M=i+8*laneHi
          float v = acc[mi][ni][i] + bias[ni];
          v = v >= 0.f ? v : NEG_SLOPE * v;
          fin[mi][ni][i] += gT[rloc * E_ + e] * v;
        }
  }

  #pragma unroll
  for (int mi = 0; mi < 2; ++mi)
    #pragma unroll
    for (int ni = 0; ni < 4; ++ni)
      #pragma unroll
      for (int i = 0; i < 8; ++i) {
        int row = m0 + waveM * 32 + mi * 16 + laneHi * 8 + i;
        int col = n0 + waveN * 64 + ni * 16 + laneLo;
        out[(size_t)row * H_ + col] = fin[mi][ni][i];
      }
}

extern "C" void kernel_launch(void* const* d_in, const int* in_sizes, int n_in,
                              void* d_out, int out_size, void* d_ws, size_t ws_size,
                              hipStream_t stream) {
  (void)in_sizes; (void)n_in; (void)out_size; (void)ws_size;
  const float* X  = (const float*)d_in[0];   // [NT, D]
  const float* Wg = (const float*)d_in[1];   // [E, D]
  const float* bg = (const float*)d_in[2];   // [E]
  const float* We = (const float*)d_in[3];   // [E, H, D]
  const float* be = (const float*)d_in[4];   // [E, H]
  float* out = (float*)d_out;                // [NT, H]

  char* ws = (char*)d_ws;
  float*  gateP = (float*)ws;                                  // NT*E f32 (256 KB)
  size_t  gateBytes = (size_t)NT * E_ * sizeof(float);
  bf16_t* Xb  = (bf16_t*)(ws + gateBytes);                     // NT*D bf16 (16.8 MB)
  size_t  xbBytes = (size_t)NT * D_ * sizeof(bf16_t);
  bf16_t* Web = (bf16_t*)(ws + gateBytes + xbBytes);           // E*H*D bf16 (33.6 MB)

  long nX = (long)NT * D_;
  long nW = (long)E_ * H_ * D_;
  cvt_f32_bf16<<<(unsigned)(nX / 8 / 256), 256, 0, stream>>>(X, Xb, nX);
  cvt_f32_bf16<<<(unsigned)(nW / 8 / 256), 256, 0, stream>>>(We, Web, nW);
  gate_kernel<<<NT / 32, 256, 0, stream>>>(X, Wg, bg, gateP);

  dim3 grid(H_ / BN, NT / BM);   // 16 x 64
  moe_wmma<<<grid, 256, 0, stream>>>(Xb, Web, be, gateP, out);
}
